// TransformerBlock_27616639713578
// MI455X (gfx1250) — matrix operand order repaired and checked by reading, not compiled, not run on hardware
//
#include <hip/hip_runtime.h>
#include <hip/hip_bf16.h>

// ---------------------------------------------------------------------------
// Transformer block for MI455X (gfx1250): bf16 WMMA GEMMs + flash attention.
// f32 accumulation via v_wmma_f32_16x16x32_bf16; LN/softmax/GELU in fp32.
// A-tiles / K-chunks staged via the Tensor Data Mover (async DMA to LDS,
// TENSORcnt-tracked); all LDS tile buffers double-buffered.
// ---------------------------------------------------------------------------

typedef __bf16 bf16;
typedef __attribute__((ext_vector_type(16))) __bf16 v16bf;
typedef __attribute__((ext_vector_type(8)))  __bf16 bf16x8;
typedef __attribute__((ext_vector_type(4)))  __bf16 bf16x4;
typedef __attribute__((ext_vector_type(8)))  float  v8f;

#define E_   1024
#define H_   16
#define HD_  64
#define FF_  4096
#define B_   4
#define S_   2048
#define M_   (B_ * S_)   // 8192 rows

__device__ __forceinline__ v8f wmma_bf16(v16bf a, v16bf b, v8f c) {
    return __builtin_amdgcn_wmma_f32_16x16x32_bf16(
        /*neg_a=*/false, a, /*neg_b=*/false, b,
        /*c_mod=*/(short)0, c, /*reuse_a=*/false, /*reuse_b=*/false);
}

// Two contiguous 16B loads -> one 16-elem bf16 fragment (ISA 16-bit A/B
// layouts: VGPRs 0-3 = first 8 elems, VGPRs 4-7 = second 8 elems).
__device__ __forceinline__ v16bf load_frag2(const bf16* p0, const bf16* p1) {
    union { v16bf v; bf16x8 h[2]; } u;
    u.h[0] = *(const bf16x8*)p0;
    u.h[1] = *(const bf16x8*)p1;
    return u.v;
}

// ---------------------------------------------------------------------------
// Tensor Data Mover: 2-D bf16 tile, global -> LDS, with LDS row padding.
// pad_interval_code: 0=2,1=4,2=8,3=16,4=32... DWORDs between pads
// pad_amount_code:   code+1 DWORDs of padding
// ---------------------------------------------------------------------------
#if __has_builtin(__builtin_amdgcn_tensor_load_to_lds)
#define HAVE_TDM 1
typedef __attribute__((ext_vector_type(4))) unsigned int u32x4;
typedef __attribute__((ext_vector_type(8))) int i32x8;
typedef __attribute__((ext_vector_type(4))) int i32x4;

__device__ __forceinline__ void tdm_load_tile_2d_bf16(
    unsigned lds_addr, const void* gsrc,
    unsigned tile_d0, unsigned tile_d1,
    unsigned long long stride_elems,
    unsigned pad_interval_code, unsigned pad_amount_code) {
    unsigned long long ga = (unsigned long long)(uintptr_t)gsrc;
    u32x4 g0;
    g0.x = 1u;                                             // count=1, user D#
    g0.y = lds_addr;                                       // LDS byte addr
    g0.z = (unsigned)ga;                                   // global addr lo
    g0.w = (unsigned)((ga >> 32) & 0x01FFFFFFull) | (2u << 30); // hi + type=2
    const unsigned td0 = 1u << 20, td1 = 1u << 20;         // no OOB clamping
    i32x8 g1;
    g1[0] = (int)((1u << 16) |                             // data_size = 2B
                  (1u << 20) |                             // pad_enable
                  (pad_interval_code << 22) |
                  (pad_amount_code << 25));
    g1[1] = (int)((td0 & 0xFFFFu) << 16);                  // tensor_dim0 lo16
    g1[2] = (int)(((td0 >> 16) & 0xFFFFu) | ((td1 & 0xFFFFu) << 16));
    g1[3] = (int)(((td1 >> 16) & 0xFFFFu) | ((tile_d0 & 0xFFFFu) << 16));
    g1[4] = (int)(tile_d1 & 0xFFFFu);                      // tile_dim2 = 0
    g1[5] = (int)(unsigned)(stride_elems & 0xFFFFFFFFull); // dim0 stride lo
    g1[6] = (int)(unsigned)((stride_elems >> 32) & 0xFFFFull);
    g1[7] = 0;                                             // dim1 stride hi
    i32x4 gz; gz[0] = gz[1] = gz[2] = gz[3] = 0;
#if defined(__clang_major__) && (__clang_major__ >= 23)
    i32x8 gz8;
    for (int i = 0; i < 8; ++i) gz8[i] = 0;
    __builtin_amdgcn_tensor_load_to_lds(g0, g1, gz, gz, gz8, 0);
#else
    __builtin_amdgcn_tensor_load_to_lds(g0, g1, gz, gz, 0);
#endif
}
#else
#define HAVE_TDM 0
#endif

// ---------------------------------------------------------------------------
// fp32 -> bf16 conversion (weights)
// ---------------------------------------------------------------------------
__global__ void cvt_f32_bf16(const float* __restrict__ in,
                             bf16* __restrict__ out, int n) {
    int i = blockIdx.x * blockDim.x + threadIdx.x;
    int stride = gridDim.x * blockDim.x;
    for (; i < n; i += stride) out[i] = (bf16)in[i];
}

// ---------------------------------------------------------------------------
// LayerNorm over E=1024, one block (256 thr) per row. Writes bf16 (+opt fp32).
// ---------------------------------------------------------------------------
__global__ __launch_bounds__(256) void layernorm_kernel(
    const float* __restrict__ x, const float* __restrict__ g,
    const float* __restrict__ bb, bf16* __restrict__ obf,
    float* __restrict__ of32) {
    const int row = blockIdx.x;
    const float* xr = x + (size_t)row * E_;
    const int tid = threadIdx.x;
    float v[4];
    float s = 0.f, ss = 0.f;
    for (int i = 0; i < 4; ++i) {
        v[i] = xr[tid + i * 256];
        s += v[i]; ss += v[i] * v[i];
    }
    for (int m = 16; m >= 1; m >>= 1) {
        s  += __shfl_xor(s,  m, 32);
        ss += __shfl_xor(ss, m, 32);
    }
    __shared__ float ws[8], wss[8];
    const int wid = tid >> 5, lane = tid & 31;
    if (lane == 0) { ws[wid] = s; wss[wid] = ss; }
    __syncthreads();
    if (tid == 0) {
        float ts = 0.f, tss = 0.f;
        for (int i = 0; i < 8; ++i) { ts += ws[i]; tss += wss[i]; }
        ws[0] = ts; wss[0] = tss;
    }
    __syncthreads();
    const float mean = ws[0] * (1.0f / E_);
    const float var  = wss[0] * (1.0f / E_) - mean * mean;
    const float rstd = rsqrtf(var + 1e-5f);
    for (int i = 0; i < 4; ++i) {
        int c = tid + i * 256;
        float y = (v[i] - mean) * rstd * g[c] + bb[c];
        obf[(size_t)row * E_ + c] = (bf16)y;
        if (of32) of32[(size_t)row * E_ + c] = y;
    }
}

// ---------------------------------------------------------------------------
// bf16 GEMM: C[M,N] = A[M,K] @ B[K,N] (both row-major), f32 accum.
// Block tile 128x128, K-step 32, 256 thr = 8 waves, wave tile 64x32.
// Double-buffered LDS; A tile via TDM (padded layout), B tile via packed-b32
// transpose stores.
// mode 0: C = acc                -> bf16
// mode 1: C = acc + bias + resid -> f32
// mode 2: C = gelu(acc + bias)   -> bf16
// ---------------------------------------------------------------------------
__global__ __launch_bounds__(256) void gemm_bf16_kernel(
    const bf16* __restrict__ A, int lda,
    const bf16* __restrict__ Bw, int ldb,
    void* __restrict__ Cout, int ldc,
    const float* __restrict__ bias,
    const float* __restrict__ resid,
    int K, int mode) {
    __shared__ bf16 As[2][128][40];   // row-major, +8 pad (TDM-compatible)
    __shared__ bf16 Bs[2][128][40];   // TRANSPOSED: Bs[n][k], +8 pad

    const int tid   = threadIdx.x;
    const int lane  = tid & 31, wid = tid >> 5;
    const int lhalf = lane & 15, hi = lane >> 4;
    const int wm = wid >> 2, wn = wid & 3;          // 2x4 wave grid
    const long rowBase = (long)blockIdx.y * 128;
    const long colBase = (long)blockIdx.x * 128;

    v8f acc[4][2];
    for (int i = 0; i < 4; ++i)
        for (int j = 0; j < 2; ++j)
            for (int e = 0; e < 8; ++e) acc[i][j][e] = 0.f;

    const int kb  = hi * 8;    // A-frag K sub-offset per lane-half
    const int kb2 = hi * 16;   // unused; B-frags also use kb
    const int nsteps = K >> 5;

    // B tile: each thread owns K-row pair (2p, 2p+1) x 8 cols; two bf16 from
    // adjacent K-rows pack into one u32 -> packed b32 (2-addr) LDS stores.
    auto loadB = [&](int k0, int buf) {
        const int p = tid >> 4;            // 0..15
        const int c = (tid & 15) * 8;      // 0..120
        const bf16* gb = Bw + (long)(k0 + 2 * p) * ldb + colBase + c;
        bf16x8 b0 = *(const bf16x8*)gb;
        bf16x8 b1 = *(const bf16x8*)(gb + ldb);
        for (int j = 0; j < 8; ++j) {
            union { bf16 h[2]; unsigned u; } pk;
            pk.h[0] = b0[j]; pk.h[1] = b1[j];
            *(unsigned*)&Bs[buf][c + j][2 * p] = pk.u;
        }
    };
    auto loadA = [&](int k0, int buf) {
#if HAVE_TDM
        if (wid == 0) {   // one DMA per block; EXEC-independent, TENSORcnt
            tdm_load_tile_2d_bf16((unsigned)(uintptr_t)&As[buf][0][0],
                                  A + rowBase * lda + k0,
                                  /*tile_d0=*/32, /*tile_d1=*/128,
                                  (unsigned long long)lda,
                                  /*interval: 16 DW*/3, /*pad: 4 DW*/3);
        }
#else
        const int r = tid >> 2;            // 0..63
        const int c = (tid & 3) * 8;       // 0,8,16,24
        const bf16* ga = A + (rowBase + r) * (long)lda + k0 + c;
        *(bf16x8*)&As[buf][r][c]      = *(const bf16x8*)ga;
        *(bf16x8*)&As[buf][r + 64][c] = *(const bf16x8*)(ga + 64L * lda);
#endif
    };

    loadA(0, 0);
    loadB(0, 0);
#if HAVE_TDM
    __builtin_amdgcn_s_wait_tensorcnt(0);
#endif
    __syncthreads();

    for (int s = 0; s < nsteps; ++s) {
        const int cur = s & 1;
        if (s + 1 < nsteps) {              // prefetch next tile (other buffer)
            loadA((s + 1) << 5, cur ^ 1);
            loadB((s + 1) << 5, cur ^ 1);
        }
        v16bf bfrag[2];
        for (int ni = 0; ni < 2; ++ni) {
            int n = wn * 32 + ni * 16 + lhalf;
            bfrag[ni] = load_frag2(&Bs[cur][n][kb], &Bs[cur][n][16 + kb]);
        }
        for (int mi = 0; mi < 4; ++mi) {
            int r = wm * 64 + mi * 16 + lhalf;
            v16bf afrag = load_frag2(&As[cur][r][kb], &As[cur][r][16 + kb]);
            for (int ni = 0; ni < 2; ++ni)
                acc[mi][ni] = wmma_bf16(afrag, bfrag[ni], acc[mi][ni]);
        }
#if HAVE_TDM
        __builtin_amdgcn_s_wait_tensorcnt(0);
#endif
        __syncthreads();
    }

    // Epilogue: C layout = row (r + 8*hi), col = lane&15 per 16x16 tile.
    for (int mi = 0; mi < 4; ++mi)
        for (int ni = 0; ni < 2; ++ni) {
            long col = colBase + wn * 32 + ni * 16 + lhalf;
            for (int r = 0; r < 8; ++r) {
                long row = rowBase + wm * 64 + mi * 16 + r + 8 * hi;
                float v = acc[mi][ni][r];
                if (mode == 0) {
                    ((bf16*)Cout)[row * ldc + col] = (bf16)v;
                } else if (mode == 1) {
                    v += bias[col] + resid[row * (long)ldc + col];
                    ((float*)Cout)[row * (long)ldc + col] = v;
                } else {
                    v += bias[col];
                    v = 0.5f * v * (1.0f + erff(v * 0.70710678118f));
                    ((bf16*)Cout)[row * ldc + col] = (bf16)v;
                }
            }
        }
}

// ---------------------------------------------------------------------------
// Flash attention. Grid (S/128, H, B); 8 waves/block, 16 query rows per wave.
// Q fragments resident in VGPRs; 32-key chunks double-buffered: K via TDM
// (padded rows), V transposed with packed b32 stores — chunk c+1 staged while
// chunk c computes. Online softmax with half-wave shfl_xor row reductions.
// ---------------------------------------------------------------------------
__global__ __launch_bounds__(256) void attn_kernel(
    const bf16* __restrict__ Qm, const bf16* __restrict__ Km,
    const bf16* __restrict__ Vm, const int* __restrict__ mask,
    bf16* __restrict__ Om) {
    __shared__ bf16 Ks[2][32][72];    // key-major (natural) -> QK^T B-frags
    __shared__ bf16 Vs[2][64][40];    // dim-major (transposed) -> P@V B-frags
    __shared__ bf16 Ps[8][16][40];    // per-wave P staging

    const int tid   = threadIdx.x;
    const int lane  = tid & 31, wid = tid >> 5;
    const int lhalf = lane & 15, hi = lane >> 4;
    const int b = blockIdx.z, h = blockIdx.y;
    const long q0 = (long)blockIdx.x * 128 + wid * 16;

    const int kb  = hi * 8;
    const int kb2 = hi * 16;

    // Resident Q fragments (16 rows x 64 dims = two 16x32 A-frags)
    const long qrow = q0 + lhalf;
    const bf16* qp = Qm + ((long)b * S_ + qrow) * E_ + h * HD_;
    const v16bf Qf0 = load_frag2(qp + kb,      qp + 16 + kb);
    const v16bf Qf1 = load_frag2(qp + 32 + kb, qp + 48 + kb);

    float mrow[8], lrow[8];
    v8f O[4];
    for (int r = 0; r < 8; ++r) { mrow[r] = -3.0e38f; lrow[r] = 0.f; }
    for (int t = 0; t < 4; ++t)
        for (int e = 0; e < 8; ++e) O[t][e] = 0.f;
    v8f zero8;
    for (int e = 0; e < 8; ++e) zero8[e] = 0.f;

    auto loadK = [&](int key0, int buf) {
#if HAVE_TDM
        if (wid == 0) {   // 32 keys x 64 dims, rows 32 DW + 4 DW pad
            tdm_load_tile_2d_bf16((unsigned)(uintptr_t)&Ks[buf][0][0],
                                  Km + ((long)b * S_ + key0) * E_ + h * HD_,
                                  /*tile_d0=*/64, /*tile_d1=*/32,
                                  (unsigned long long)E_,
                                  /*interval: 32 DW*/4, /*pad: 4 DW*/3);
        }
#else
        const int kk = tid >> 3;         // 0..31 (key within chunk)
        const int d  = (tid & 7) * 8;    // 0..56 (dim)
        const bf16* kg = Km + ((long)b * S_ + key0 + kk) * E_ + h * HD_ + d;
        *(bf16x8*)&Ks[buf][kk][d] = *(const bf16x8*)kg;
#endif
    };
    auto loadV = [&](int key0, int buf) {
        // key pair (2p,2p+1) x 4 dims -> 4 packed b32 transposed stores
        const int p  = tid & 15;
        const int d4 = (tid >> 4) * 4;
        const bf16* vg = Vm + ((long)b * S_ + key0 + 2 * p) * E_ + h * HD_ + d4;
        bf16x4 v0 = *(const bf16x4*)vg;
        bf16x4 v1 = *(const bf16x4*)(vg + E_);
        for (int j = 0; j < 4; ++j) {
            union { bf16 hh[2]; unsigned u; } pk;
            pk.hh[0] = v0[j]; pk.hh[1] = v1[j];
            *(unsigned*)&Vs[buf][d4 + j][2 * p] = pk.u;
        }
    };

    loadK(0, 0);
    loadV(0, 0);
#if HAVE_TDM
    __builtin_amdgcn_s_wait_tensorcnt(0);
#endif
    __syncthreads();

    const int nch = S_ / 32;
    for (int c = 0; c < nch; ++c) {
        const int cur = c & 1;
        const int key0 = c * 32;
        if (c + 1 < nch) {                 // stage next chunk (other buffer)
            loadK(key0 + 32, cur ^ 1);
            loadV(key0 + 32, cur ^ 1);
        }

        // S = Q @ K^T for two 16-key subtiles (K-dim 64 = 2 WMMAs each)
        v8f S0, S1;
        {
            v16bf Bk0 = load_frag2(&Ks[cur][lhalf][kb],
                                   &Ks[cur][lhalf][16 + kb]);
            v16bf Bk1 = load_frag2(&Ks[cur][lhalf][32 + kb],
                                   &Ks[cur][lhalf][48 + kb]);
            S0 = wmma_bf16(Qf0, Bk0, zero8);
            S0 = wmma_bf16(Qf1, Bk1, S0);
            v16bf Bk2 = load_frag2(&Ks[cur][16 + lhalf][kb],
                                   &Ks[cur][16 + lhalf][16 + kb]);
            v16bf Bk3 = load_frag2(&Ks[cur][16 + lhalf][32 + kb],
                                   &Ks[cur][16 + lhalf][48 + kb]);
            S1 = wmma_bf16(Qf0, Bk2, zero8);
            S1 = wmma_bf16(Qf1, Bk3, S1);
        }
        // scale 1/sqrt(64) + mask (mask indexed per key)
        const int mk0 = mask[b * S_ + key0 + lhalf];
        const int mk1 = mask[b * S_ + key0 + 16 + lhalf];
        for (int r = 0; r < 8; ++r) {
            S0[r] = (mk0 == 0) ? -1.0e20f : S0[r] * 0.125f;
            S1[r] = (mk1 == 0) ? -1.0e20f : S1[r] * 0.125f;
        }
        // Online softmax (rows live in 16-lane halves -> xor masks 8..1)
        float pr0[8], pr1[8];
        for (int r = 0; r < 8; ++r) {
            float mx = fmaxf(S0[r], S1[r]);
            for (int mm = 8; mm >= 1; mm >>= 1)
                mx = fmaxf(mx, __shfl_xor(mx, mm, 32));
            float mnew  = fmaxf(mrow[r], mx);
            float alpha = __expf(mrow[r] - mnew);
            mrow[r] = mnew;
            lrow[r] *= alpha;
            for (int t = 0; t < 4; ++t) O[t][r] *= alpha;
            float p0 = __expf(S0[r] - mnew);
            float p1 = __expf(S1[r] - mnew);
            float psum = p0 + p1;
            for (int mm = 8; mm >= 1; mm >>= 1)
                psum += __shfl_xor(psum, mm, 32);
            lrow[r] += psum;
            pr0[r] = p0; pr1[r] = p1;
        }
        // Stage P (C layout) into per-wave LDS, re-read as A-frag layout.
        for (int r = 0; r < 8; ++r) {
            Ps[wid][r + 8 * hi][lhalf]      = (bf16)pr0[r];
            Ps[wid][r + 8 * hi][16 + lhalf] = (bf16)pr1[r];
        }
        asm volatile("s_wait_dscnt 0x0" ::: "memory");  // intra-wave LDS RAW
        v16bf Pf = load_frag2(&Ps[wid][lhalf][kb], &Ps[wid][lhalf][16 + kb]);
        // O += P @ V  (4 dim tiles of 16)
        for (int t = 0; t < 4; ++t) {
            v16bf Vf = load_frag2(&Vs[cur][t * 16 + lhalf][kb],
                                  &Vs[cur][t * 16 + lhalf][16 + kb]);
            O[t] = wmma_bf16(Pf, Vf, O[t]);
        }
#if HAVE_TDM
        __builtin_amdgcn_s_wait_tensorcnt(0);  // next-chunk DMA landed
#endif
        __syncthreads();
    }

    // Normalize and write O (bf16) back into [B,S,H,HD] layout.
    bf16* op = Om + ((long)b * S_ + q0) * E_ + h * HD_;
    for (int r = 0; r < 8; ++r) {
        float inv = 1.0f / lrow[r];
        long row = r + 8 * hi;
        for (int t = 0; t < 4; ++t)
            op[row * E_ + t * 16 + lhalf] = (bf16)(O[t][r] * inv);
    }
}

// ---------------------------------------------------------------------------
// Host-side launch sequence (graph-capture safe: only kernel launches).
// Workspace requirement: ~280 MB.
// ---------------------------------------------------------------------------
extern "C" void kernel_launch(void* const* d_in, const int* in_sizes, int n_in,
                              void* d_out, int out_size, void* d_ws,
                              size_t ws_size, hipStream_t stream) {
    const float* value = (const float*)d_in[0];
    const float* key   = (const float*)d_in[1];
    const float* query = (const float*)d_in[2];
    const int*   mask  = (const int*)  d_in[3];
    const float* Wv    = (const float*)d_in[4];
    const float* Wk    = (const float*)d_in[5];
    const float* Wq    = (const float*)d_in[6];
    const float* Wo    = (const float*)d_in[7];
    const float* bo    = (const float*)d_in[8];
    const float* ln1_g = (const float*)d_in[9];
    const float* ln1_b = (const float*)d_in[10];
    const float* ln2_g = (const float*)d_in[11];
    const float* ln2_b = (const float*)d_in[12];
    const float* lnf_g = (const float*)d_in[13];
    const float* lnf_b = (const float*)d_in[14];
    const float* W1    = (const float*)d_in[15];
    const float* b1    = (const float*)d_in[16];
    const float* W2    = (const float*)d_in[17];
    const float* b2    = (const float*)d_in[18];

    char* w = (char*)d_ws;
    auto alloc = [&](size_t bytes) -> void* {
        void* p = (void*)w;
        w += (bytes + 255) & ~(size_t)255;
        return p;
    };
    bf16* Wv_b  = (bf16*)alloc((size_t)E_ * E_ * 2);
    bf16* Wk_b  = (bf16*)alloc((size_t)E_ * E_ * 2);
    bf16* Wq_b  = (bf16*)alloc((size_t)E_ * E_ * 2);
    bf16* Wo_b  = (bf16*)alloc((size_t)E_ * E_ * 2);
    bf16* W1_b  = (bf16*)alloc((size_t)E_ * FF_ * 2);
    bf16* W2_b  = (bf16*)alloc((size_t)FF_ * E_ * 2);
    bf16* v_b   = (bf16*)alloc((size_t)M_ * E_ * 2);
    bf16* k_b   = (bf16*)alloc((size_t)M_ * E_ * 2);
    bf16* q_b   = (bf16*)alloc((size_t)M_ * E_ * 2);
    float* q_f  = (float*)alloc((size_t)M_ * E_ * 4);   // fp32 q for residual
    bf16* Qb    = (bf16*)alloc((size_t)M_ * E_ * 2);
    bf16* Kb    = (bf16*)alloc((size_t)M_ * E_ * 2);
    bf16* Vb    = (bf16*)alloc((size_t)M_ * E_ * 2);
    bf16* attnb = (bf16*)alloc((size_t)M_ * E_ * 2);
    float* x_f  = (float*)alloc((size_t)M_ * E_ * 4);   // attn out + residual
    bf16* xn_b  = (bf16*)alloc((size_t)M_ * E_ * 2);
    bf16* f1_b  = (bf16*)alloc((size_t)M_ * FF_ * 2);

    // 1) weight down-conversion
    cvt_f32_bf16<<<1024, 256, 0, stream>>>(Wv, Wv_b, E_ * E_);
    cvt_f32_bf16<<<1024, 256, 0, stream>>>(Wk, Wk_b, E_ * E_);
    cvt_f32_bf16<<<1024, 256, 0, stream>>>(Wq, Wq_b, E_ * E_);
    cvt_f32_bf16<<<1024, 256, 0, stream>>>(Wo, Wo_b, E_ * E_);
    cvt_f32_bf16<<<4096, 256, 0, stream>>>(W1, W1_b, E_ * FF_);
    cvt_f32_bf16<<<4096, 256, 0, stream>>>(W2, W2_b, FF_ * E_);

    // 2) pre-norms
    layernorm_kernel<<<M_, 256, 0, stream>>>(value, ln2_g, ln2_b, v_b, nullptr);
    layernorm_kernel<<<M_, 256, 0, stream>>>(key,   ln2_g, ln2_b, k_b, nullptr);
    layernorm_kernel<<<M_, 256, 0, stream>>>(query, ln1_g, ln1_b, q_b, q_f);

    // 3) QKV projections (bf16 out)
    dim3 gE(E_ / 128, M_ / 128);
    gemm_bf16_kernel<<<gE, 256, 0, stream>>>(q_b, E_, Wq_b, E_, Qb, E_,
                                             nullptr, nullptr, E_, 0);
    gemm_bf16_kernel<<<gE, 256, 0, stream>>>(k_b, E_, Wk_b, E_, Kb, E_,
                                             nullptr, nullptr, E_, 0);
    gemm_bf16_kernel<<<gE, 256, 0, stream>>>(v_b, E_, Wv_b, E_, Vb, E_,
                                             nullptr, nullptr, E_, 0);

    // 4) flash attention
    dim3 ga(S_ / 128, H_, B_);
    attn_kernel<<<ga, 256, 0, stream>>>(Qb, Kb, Vb, mask, attnb);

    // 5) output projection + bias + residual(q) -> fp32 x
    gemm_bf16_kernel<<<gE, 256, 0, stream>>>(attnb, E_, Wo_b, E_, x_f, E_,
                                             bo, q_f, E_, 1);

    // 6) final norm
    layernorm_kernel<<<M_, 256, 0, stream>>>(x_f, lnf_g, lnf_b, xn_b, nullptr);

    // 7) FFN up: gelu(xn @ W1 + b1) -> bf16
    dim3 gF(FF_ / 128, M_ / 128);
    gemm_bf16_kernel<<<gF, 256, 0, stream>>>(xn_b, E_, W1_b, FF_, f1_b, FF_,
                                             b1, nullptr, E_, 2);

    // 8) FFN down + bias + residual(x) -> fp32 d_out
    gemm_bf16_kernel<<<gE, 256, 0, stream>>>(f1_b, FF_, W2_b, E_,
                                             (float*)d_out, E_,
                                             b2, x_f, FF_, 1);
}
